// MultiLinearModel_69097433858169
// MI455X (gfx1250) — matrix operand order repaired and checked by reading, not compiled, not run on hardware
//
#include <hip/hip_runtime.h>

// ---------------------------------------------------------------------------
// Ternary 3-layer MLP for MI455X (gfx1250), bf16 WMMA path.
//   h1 = relu(x @ T(W1)^T + b1); h2 = relu(h1 @ T(W2)^T + b2); out = h2 @ T(W3)^T + b3
// Ternary weights are exact in bf16; activations bf16, fp32 accumulate in WMMA.
// Double-buffered LDS staging via async global->LDS DMA (ASYNCcnt) when the
// toolchain exposes the gfx1250 builtin; sync fallback otherwise.
// ---------------------------------------------------------------------------

typedef __attribute__((ext_vector_type(16))) __bf16 v16bf;
typedef __attribute__((ext_vector_type(8)))  float  v8f;

#define HAS_ASYNC_LDS __has_builtin(__builtin_amdgcn_global_load_async_to_lds_b128)

// Builtin prototype (from hipcc diagnostic):
//   void __builtin_amdgcn_global_load_async_to_lds_b128(
//       v4i __attribute__((address_space(1)))* src,   // prints as "__device__"
//       v4i __attribute__((address_space(3)))* dst,
//       int offset, int cpol)
typedef int v4i __attribute__((vector_size(16)));
typedef v4i __attribute__((address_space(1))) gl_v4i;
typedef v4i __attribute__((address_space(3))) lds_v4i;

union Frag {
    uint4 u4[2];
    v16bf v;
};

__device__ __forceinline__ unsigned short f32_to_bf16_bits(float f) {
    unsigned u = __float_as_uint(f);
    unsigned r = u + 0x7FFFu + ((u >> 16) & 1u);   // round-to-nearest-even
    return (unsigned short)(r >> 16);
}

// ---- preprocessing ---------------------------------------------------------

// {-1,0,+1} threshold quantization straight to bf16 bit patterns (lossless).
__global__ void ternary_quant_kernel(const float* __restrict__ W,
                                     unsigned short* __restrict__ Wq, long n) {
    long i = (long)blockIdx.x * blockDim.x + threadIdx.x;
    long stride = (long)gridDim.x * blockDim.x;
    for (; i < n; i += stride) {
        float w = W[i];
        unsigned short q = 0;
        if (fabsf(w) > 0.7f) q = (w > 0.f) ? (unsigned short)0x3F80u
                                           : (unsigned short)0xBF80u;
        Wq[i] = q;
    }
}

__global__ void f32_to_bf16_kernel(const float* __restrict__ in,
                                   unsigned short* __restrict__ out, long n) {
    long i = (long)blockIdx.x * blockDim.x + threadIdx.x;
    long stride = (long)gridDim.x * blockDim.x;
    for (; i < n; i += stride) out[i] = f32_to_bf16_bits(in[i]);
}

// ---- fused GEMM + bias (+ReLU) --------------------------------------------
// Y[M,N] = act(X[M,K] @ W[N,K]^T + bias), X/W bf16 (as ushort bits).
// Workgroup: 128x128 tile, 4 waves, each wave a 64x64 sub-tile.

#define TILE       128
#define KSTEP      64
#define LDS_STRIDE 72   // bf16 elems; 144 B row stride: 16B-aligned, bank-spread

template <bool RELU, bool OUT_BF16>
__global__ __launch_bounds__(128) void gemm_tern(
    const unsigned short* __restrict__ X,   // [M,K] bf16 bits
    const unsigned short* __restrict__ W,   // [N,K] bf16 bits (ternary)
    const float* __restrict__ bias,         // [N]
    void* __restrict__ Yout,                // [M,N] bf16 bits or float
    int M, int N, int K)
{
    __shared__ unsigned short Xs[2][TILE * LDS_STRIDE];
    __shared__ unsigned short Ws[2][TILE * LDS_STRIDE];

    const int tid  = threadIdx.x;
    const int lane = tid & 31;
    const int wave = tid >> 5;      // 0..3
    const int wrow = wave >> 1;     // 0..1 : 64-row half of the block tile
    const int wcol = wave & 1;      // 0..1 : 64-col half
    const int n16  = lane & 15;
    const int hi   = lane >> 4;

    const int bt = blockIdx.y * TILE;   // M base
    const int ot = blockIdx.x * TILE;   // N base

    // staging map: 128 threads -> 16 rows x (8 lanes * 8 elems) per pass
    const int srow = tid >> 3;          // 0..15
    const int scol = (tid & 7) * 8;     // 0..56, 16B chunks

    // issue one K-stage (TILE x KSTEP of X and W) into LDS buffer `buf`
    auto stage = [&](int buf, int k0) {
        #pragma unroll
        for (int p = 0; p < 8; p++) {
            const int r = srow + 16 * p;
            const size_t xoff = (size_t)(bt + r) * K + k0 + scol;
            const size_t woff = (size_t)(ot + r) * K + k0 + scol;
            unsigned short* xl = &Xs[buf][r * LDS_STRIDE + scol];
            unsigned short* wl = &Ws[buf][r * LDS_STRIDE + scol];
#if HAS_ASYNC_LDS
            // gfx1250 async DMA: global -> LDS directly, tracked by ASYNCcnt
            __builtin_amdgcn_global_load_async_to_lds_b128(
                (gl_v4i*)(X + xoff), (lds_v4i*)xl, 0, 0);
            __builtin_amdgcn_global_load_async_to_lds_b128(
                (gl_v4i*)(W + woff), (lds_v4i*)wl, 0, 0);
#else
            uint4 xv = *(const uint4*)(X + xoff);
            uint4 wv = *(const uint4*)(W + woff);
            *(uint4*)xl = xv;
            *(uint4*)wl = wv;
#endif
        }
    };

    auto stage_wait = [&]() {
#if HAS_ASYNC_LDS
  #if __has_builtin(__builtin_amdgcn_s_wait_asynccnt)
        __builtin_amdgcn_s_wait_asynccnt(0);
  #else
        asm volatile("s_wait_asynccnt 0x0" ::: "memory");
  #endif
#endif
        __syncthreads();
    };

    v8f acc[4][4] = {};

    stage(0, 0);
    stage_wait();

    const int nsteps = K / KSTEP;
    for (int s = 0; s < nsteps; s++) {
        const int cur = s & 1;
        if (s + 1 < nsteps) stage(cur ^ 1, (s + 1) * KSTEP);   // prefetch stage

        const unsigned short* xb = &Xs[cur][0];
        const unsigned short* wb = &Ws[cur][0];

        #pragma unroll
        for (int kk = 0; kk < KSTEP; kk += 32) {
            // B-frags (32x16 bf16): lane holds 16 K-contiguous elems at k=16*hi
            Frag bfr[4];
            #pragma unroll
            for (int nt = 0; nt < 4; nt++) {
                const unsigned short* bp =
                    wb + (wcol * 64 + nt * 16 + n16) * LDS_STRIDE + kk + 16 * hi;
                bfr[nt].u4[0] = *(const uint4*)(bp);
                bfr[nt].u4[1] = *(const uint4*)(bp + 8);
            }
            #pragma unroll
            for (int mt = 0; mt < 4; mt++) {
                // A-frag (16x32 bf16): lane holds 16 K-contiguous elems at k=16*hi
                Frag afr;
                const unsigned short* ap =
                    xb + (wrow * 64 + mt * 16 + n16) * LDS_STRIDE + kk;
                afr.u4[0] = *(const uint4*)(ap + 16 * hi);
                afr.u4[1] = *(const uint4*)(ap + 16 * hi + 8);
                #pragma unroll
                for (int nt = 0; nt < 4; nt++) {
                    acc[mt][nt] = __builtin_amdgcn_wmma_f32_16x16x32_bf16(
                        false, afr.v, false, bfr[nt].v,
                        (short)0, acc[mt][nt], false, false);
                }
            }
        }
        stage_wait();   // next buffer resident + all waves done reading `cur`
    }

    // epilogue: C/D layout -> lane n16 = column, VGPR r -> row r + 8*hi
    #pragma unroll
    for (int nt = 0; nt < 4; nt++) {
        const int o  = ot + wcol * 64 + nt * 16 + n16;
        const float bo = bias[o];
        #pragma unroll
        for (int mt = 0; mt < 4; mt++) {
            #pragma unroll
            for (int r = 0; r < 8; r++) {
                float v = acc[mt][nt][r] + bo;
                if (RELU) v = v > 0.f ? v : 0.f;
                const int row = bt + wrow * 64 + mt * 16 + hi * 8 + r;
                if (OUT_BF16) {
                    ((unsigned short*)Yout)[(size_t)row * N + o] = f32_to_bf16_bits(v);
                } else {
                    ((float*)Yout)[(size_t)row * N + o] = v;
                }
            }
        }
    }
}

// ---------------------------------------------------------------------------

extern "C" void kernel_launch(void* const* d_in, const int* in_sizes, int n_in,
                              void* d_out, int out_size, void* d_ws, size_t ws_size,
                              hipStream_t stream) {
    (void)in_sizes; (void)n_in; (void)out_size; (void)ws_size;

    const float* x  = (const float*)d_in[0];
    const float* W1 = (const float*)d_in[1];
    const float* b1 = (const float*)d_in[2];
    const float* W2 = (const float*)d_in[3];
    const float* b2 = (const float*)d_in[4];
    const float* W3 = (const float*)d_in[5];
    const float* b3 = (const float*)d_in[6];

    const int B = 8192, D0 = 4096, D1 = 4096, D2 = 4096, D3 = 2048;

    // workspace layout (bf16 bits). H2 reuses Xq (dead after layer 1). ~218 MB.
    char* ws = (char*)d_ws;
    unsigned short* Xq  = (unsigned short*)ws;                  // B*D0
    unsigned short* W1q = Xq  + (size_t)B  * D0;                // D1*D0
    unsigned short* W2q = W1q + (size_t)D1 * D0;                // D2*D1
    unsigned short* W3q = W2q + (size_t)D2 * D1;                // D3*D2
    unsigned short* H1  = W3q + (size_t)D3 * D2;                // B*D1
    unsigned short* H2  = Xq;                                   // alias: B*D2

    const int tpb = 256;
    f32_to_bf16_kernel  <<<2048, tpb, 0, stream>>>(x,  Xq,  (long)B  * D0);
    ternary_quant_kernel<<<2048, tpb, 0, stream>>>(W1, W1q, (long)D1 * D0);
    ternary_quant_kernel<<<2048, tpb, 0, stream>>>(W2, W2q, (long)D2 * D1);
    ternary_quant_kernel<<<1024, tpb, 0, stream>>>(W3, W3q, (long)D3 * D2);

    dim3 blk(128, 1, 1);
    gemm_tern<true,  true ><<<dim3(D1 / TILE, B / TILE), blk, 0, stream>>>(
        Xq, W1q, b1, (void*)H1, B, D1, D0);
    gemm_tern<true,  true ><<<dim3(D2 / TILE, B / TILE), blk, 0, stream>>>(
        H1, W2q, b2, (void*)H2, B, D2, D1);
    gemm_tern<false, false><<<dim3(D3 / TILE, B / TILE), blk, 0, stream>>>(
        H2, W3q, b3, d_out, B, D3, D2);
}
